// PermInvConvModel_68118181315149
// MI455X (gfx1250) — matrix operand order repaired and checked by reading, not compiled, not run on hardware
//
#include <hip/hip_runtime.h>
#include <math.h>

// ---------------------------------------------------------------------------
// PermInvConvModel for MI455X (gfx1250, wave32, WMMA)
//
//   cvt_w1    : transpose W1[:F] (f,n) -> two bf16 planes (n,f) hi/lo with
//               f zero-padded to 32; B-fragments become 2x global_load_b128.
//   [i>0] zero_pt; prev_gemm (f32 WMMA 16x16x4, K-split + atomics) -> pt
//   gemm1_ss  : Zraw = p @ W1[:F] via bf16x3 split WMMA 16x16x32 (fp32-class
//               accuracy), accumulates sum(p^2) -> sfacc.  Single pass over p.
//   finalize  : sf = sqrt(sfacc); mult += log(sf); sfinv = 1/sf
//   stage2    : A = relu(Zraw/sf + b1 + pt); H = relu(A @ W2 + b2) via f32
//               WMMA 16x16x4; rep += sum_c H (atomics); fp = H flattened
//   final_out : out[b] = concat(reps, mult) @ Wr + br
// ---------------------------------------------------------------------------

typedef __attribute__((ext_vector_type(2)))  float   v2f;
typedef __attribute__((ext_vector_type(8)))  float   v8f;
typedef __attribute__((ext_vector_type(8)))  __bf16  v8bf;
typedef __attribute__((ext_vector_type(16))) __bf16  v16bf;

#define NB 16
#define H1 128
#define H2 64

union AFrag  { v16bf v; __bf16 b[16]; };
union BFrag  { v16bf v; v8bf   h[2];  };

__device__ __forceinline__ v8f wmma_f32x4(v2f a, v2f b, v8f c) {
  return __builtin_amdgcn_wmma_f32_16x16x4_f32(false, a, false, b, (short)0, c, false, false);
}
__device__ __forceinline__ v8f wmma_bf16(v16bf a, v16bf b, v8f c) {
  return __builtin_amdgcn_wmma_f32_16x16x32_bf16(false, a, false, b, (short)0, c, false, false);
}

// ---------------- workspace init -------------------------------------------
__global__ void init_ws(float* __restrict__ sfacc, float* __restrict__ mult,
                        float* __restrict__ rep) {
  int t = threadIdx.x;
  if (t < 80) sfacc[t] = 0.0f;
  if (t < 16) mult[t] = 1.0f;           // mult starts as ones
  for (int k = t; k < 5 * NB * H2; k += 256) rep[k] = 0.0f;
}

__global__ void zero_pt(float* __restrict__ pt) {
  int i = blockIdx.x * 256 + threadIdx.x;   // 8 blocks x 256 == 2048 exactly
  pt[i] = 0.0f;
}

// ---------------- W1 split + transpose -------------------------------------
// In : W1 (F x 128) f32, row-major.
// Out: Whi/Wlo (128 x Fpad) bf16, f-contiguous, hi/lo split, tail zero-padded.
// 32x32 tile transpose through LDS; coalesced reads and writes.
__global__ __launch_bounds__(256)
void cvt_w1(const float* __restrict__ W, __bf16* __restrict__ Whi,
            __bf16* __restrict__ Wlo, int F, int Fpad) {
  __shared__ unsigned tile[32][33];
  const int tx = threadIdx.x;          // 0..31
  const int ty = threadIdx.y;          // 0..7
  const int f0 = blockIdx.x * 32;
  const int n0 = blockIdx.y * 32;
#pragma unroll
  for (int i = 0; i < 4; ++i) {
    int fr = f0 + ty + 8 * i;
    float x = (fr < F) ? W[(size_t)fr * H1 + (n0 + tx)] : 0.0f;
    __bf16 h = (__bf16)x;
    __bf16 l = (__bf16)(x - (float)h);
    unsigned uh = (unsigned)__builtin_bit_cast(unsigned short, h);
    unsigned ul = (unsigned)__builtin_bit_cast(unsigned short, l);
    tile[ty + 8 * i][tx] = uh | (ul << 16);
  }
  __syncthreads();
#pragma unroll
  for (int i = 0; i < 4; ++i) {
    int nw = n0 + ty + 8 * i;
    int fw = f0 + tx;
    unsigned u = tile[tx][ty + 8 * i];
    Whi[(size_t)nw * Fpad + fw] = __builtin_bit_cast(__bf16, (unsigned short)(u & 0xFFFFu));
    Wlo[(size_t)nw * Fpad + fw] = __builtin_bit_cast(__bf16, (unsigned short)(u >> 16));
  }
}

// ---------------- GEMM1: Zraw = p @ W1[:F]  (bf16x3) -----------------------
// A[(b,c), f] = p[b, f, c]. One wave owns a 16(M) x 128(N) strip; K stepped 32.
template <bool TAIL>
__device__ __forceinline__ void g1_kstep(const float* __restrict__ pA,
                                         const __bf16* __restrict__ Whi,
                                         int fbase, int F, int C, int Fpad,
                                         int planeStride, int kh, int mrow,
                                         float& ss, v8f* acc) {
  AFrag ahi, alo;
#pragma unroll
  for (int e = 0; e < 16; ++e) {
    // A 16-bit layout (contiguous, matches B): element e -> K = 16*kh + e
    int f = fbase + 16 * kh + e;
    float x;
    if (TAIL) {
      int fc = (f < F) ? f : (F - 1);
      x = pA[(size_t)fc * C];
      x = (f < F) ? x : 0.0f;
    } else {
      x = pA[(size_t)f * C];
    }
    ss += x * x;                      // Frobenius accumulation (raw p)
    __bf16 h = (__bf16)x;
    ahi.b[e] = h;
    alo.b[e] = (__bf16)(x - (float)h);
  }
#pragma unroll
  for (int t = 0; t < 8; ++t) {
    // B k order used here (contiguous, matches A): lanes 0-15 hold K=0..15, lanes 16-31 hold K=16..31;
    // fragment = 16 contiguous bf16 in the transposed plane -> 2x b128 loads.
    const __bf16* bp = Whi + (size_t)(t * 16 + mrow) * Fpad + fbase + 16 * kh;
    BFrag bhi, blo;
    bhi.h[0] = *(const v8bf*)(bp);
    bhi.h[1] = *(const v8bf*)(bp + 8);
    blo.h[0] = *(const v8bf*)(bp + planeStride);
    blo.h[1] = *(const v8bf*)(bp + planeStride + 8);
    acc[t] = wmma_bf16(ahi.v, bhi.v, acc[t]);   // hi*hi
    acc[t] = wmma_bf16(alo.v, bhi.v, acc[t]);   // lo*hi
    acc[t] = wmma_bf16(ahi.v, blo.v, acc[t]);   // hi*lo
  }
}

__global__ __launch_bounds__(128)
void gemm1_ss(const float* __restrict__ p, const __bf16* __restrict__ Whi,
              float* __restrict__ zraw, float* __restrict__ sfacc,
              int C, int F, int Fpad) {
  const int lane = threadIdx.x & 31;
  const int wave = threadIdx.x >> 5;
  const int strip = blockIdx.x * 4 + wave;      // grid sized exactly
  const int row0 = strip * 16;                  // row in flattened M = B*C
  const int b    = row0 / C;
  const int c0   = row0 - b * C;
  const int mrow = lane & 15;
  const int kh   = lane >> 4;
  const int planeStride = H1 * Fpad;            // Whi -> Wlo
  const float* pA = p + (size_t)b * F * C + (c0 + mrow);  // element f at pA[f*C]

  v8f acc[8] = {};
  float ss = 0.0f;
  const int fullks = F >> 5;
  for (int ks = 0; ks < fullks; ++ks)
    g1_kstep<false>(pA, Whi, ks << 5, F, C, Fpad, planeStride, kh, mrow, ss, acc);
  if (F & 31)  // B side is zero-padded; only A needs the guard
    g1_kstep<true>(pA, Whi, fullks << 5, F, C, Fpad, planeStride, kh, mrow, ss, acc);

  // wave-reduce sum of squares, one atomic per wave
  for (int off = 16; off > 0; off >>= 1) ss += __shfl_down(ss, off, 32);
  if (lane == 0) atomicAdd(&sfacc[b], ss);

  // D layout: VGPR j -> row j (lanes 0-15) / j+8 (lanes 16-31), col = lane%16
#pragma unroll
  for (int t = 0; t < 8; ++t)
#pragma unroll
    for (int j = 0; j < 8; ++j)
      zraw[(size_t)(row0 + j + 8 * kh) * H1 + t * 16 + mrow] = acc[t][j];
}

// ---------------- prev term: pt[b,h] = for_prev[b,:] @ W1[F:,h] ------------
__global__ __launch_bounds__(256)
void prev_gemm(const float* __restrict__ fp, const float* __restrict__ W1tail,
               float* __restrict__ pt, int Kd, int kchunk) {
  const int wave = threadIdx.x >> 5;     // N-tile 0..7
  const int lane = threadIdx.x & 31;
  const int mrow = lane & 15;            // b index
  const int kh   = lane >> 4;
  const int n    = wave * 16 + mrow;
  const int k0 = blockIdx.x * kchunk;
  int k1 = k0 + kchunk; if (k1 > Kd) k1 = Kd;
  v8f acc = {};
  for (int k = k0; k < k1; k += 4) {
    int d = k + 2 * kh;                  // A 32-bit layout: lane-halves own K pairs
    v2f a, bb;
    a[0]  = fp[(size_t)mrow * Kd + d];
    a[1]  = fp[(size_t)mrow * Kd + d + 1];
    bb[0] = W1tail[(size_t)d * H1 + n];
    bb[1] = W1tail[(size_t)(d + 1) * H1 + n];
    acc = wmma_f32x4(a, bb, acc);
  }
#pragma unroll
  for (int j = 0; j < 8; ++j)
    atomicAdd(&pt[(j + 8 * kh) * H1 + n], acc[j]);
}

// ---------------- finalize sf / mult ---------------------------------------
__global__ void finalize_sf(const float* __restrict__ sfacc, float* __restrict__ sfinv,
                            float* __restrict__ mult) {
  int b = threadIdx.x;
  if (b < 16) {
    float s = sqrtf(sfacc[b]);
    sfinv[b] = 1.0f / s;
    mult[b] += logf(s);
  }
}

// ---------------- stage2: scale/bias/relu + GEMM2 + relu + pool + fp -------
__global__ __launch_bounds__(128)
void stage2(const float* __restrict__ zraw, const float* __restrict__ b1,
            const float* __restrict__ W2, const float* __restrict__ b2,
            const float* __restrict__ pt, const float* __restrict__ sfinv,
            float* __restrict__ rep, float* __restrict__ fp,
            int C, int havePrev) {
  const int lane = threadIdx.x & 31;
  const int wave = threadIdx.x >> 5;
  const int strip = blockIdx.x * 4 + wave;
  const int row0 = strip * 16;
  const int b    = row0 / C;
  const int c0   = row0 - b * C;
  const int mrow = lane & 15;
  const int kh   = lane >> 4;
  const float inv = sfinv[b];

  v8f acc[4] = {};
  const float* zr = zraw + (size_t)(row0 + mrow) * H1;
  for (int k0 = 0; k0 < H1; k0 += 4) {
    v2f a;
#pragma unroll
    for (int v = 0; v < 2; ++v) {
      int h = k0 + 2 * kh + v;
      float z = zr[h] * inv + b1[h];
      if (havePrev) z += pt[b * H1 + h];
      a[v] = z > 0.0f ? z : 0.0f;        // relu -> A operand of GEMM2
    }
#pragma unroll
    for (int t = 0; t < 4; ++t) {
      v2f bb;
      bb[0] = W2[(size_t)(k0 + 2 * kh) * H2 + t * 16 + mrow];
      bb[1] = W2[(size_t)(k0 + 2 * kh + 1) * H2 + t * 16 + mrow];
      acc[t] = wmma_f32x4(a, bb, acc[t]);
    }
  }
  float* fpb = fp + (size_t)b * C * H2;
#pragma unroll
  for (int t = 0; t < 4; ++t) {
    int n = t * 16 + mrow;
    float bias = b2[n];
    float part = 0.0f;
#pragma unroll
    for (int j = 0; j < 8; ++j) {
      float h = acc[t][j] + bias;
      h = h > 0.0f ? h : 0.0f;
      fpb[(size_t)(c0 + j + 8 * kh) * H2 + n] = h;  // for_prev, c-major
      part += h;
    }
    atomicAdd(&rep[b * H2 + n], part);              // sum-pool over channels
  }
}

// ---------------- final linear ---------------------------------------------
__global__ void final_out(const float* __restrict__ rep, const float* __restrict__ mult,
                          const float* __restrict__ Wr, const float* __restrict__ br,
                          float* __restrict__ out) {
  int b = threadIdx.x;
  if (b < 16) {
    float s = 0.0f;
    for (int i = 0; i < 5; ++i)
      for (int h = 0; h < H2; ++h)
        s += rep[(i * NB + b) * H2 + h] * Wr[i * H2 + h];
    s += mult[b] * Wr[5 * H2];
    out[b] = s + br[0];
  }
}

// ---------------------------------------------------------------------------
extern "C" void kernel_launch(void* const* d_in, const int* in_sizes, int n_in,
                              void* d_out, int out_size, void* d_ws, size_t ws_size,
                              hipStream_t stream) {
  (void)in_sizes; (void)n_in; (void)out_size; (void)ws_size;
  static const int C[5] = {64, 128, 256, 512, 512};
  static const int F[5] = {36, 585, 1161, 2313, 4617};

  const float* P[5]; const float* W1[5]; const float* B1[5];
  const float* W2[5]; const float* B2[5];
  for (int i = 0; i < 5; ++i) P[i] = (const float*)d_in[i];
  for (int i = 0; i < 5; ++i) {
    W1[i] = (const float*)d_in[5 + 4 * i + 0];
    B1[i] = (const float*)d_in[5 + 4 * i + 1];
    W2[i] = (const float*)d_in[5 + 4 * i + 2];
    B2[i] = (const float*)d_in[5 + 4 * i + 3];
  }
  const float* Wr = (const float*)d_in[25];
  const float* br = (const float*)d_in[26];
  float* out = (float*)d_out;

  // workspace layout (floats): ~10.8 MB total
  float* ws    = (float*)d_ws;
  float* sfacc = ws + 0;                       // 5*16
  float* sfinv = ws + 80;                      // 5*16
  float* mult  = ws + 160;                     // 16
  float* rep   = ws + 176;                     // 5*16*64
  float* pt    = ws + 5296;                    // 16*128
  float* zraw  = ws + 8192;                    // 16*512*128 max
  float* fp    = ws + 8192 + 1048576;          // 16*512*64 max
  __bf16* wsplit = (__bf16*)(ws + 8192 + 1048576 + 524288); // 2*128*sum(Fpad)

  init_ws<<<1, 256, 0, stream>>>(sfacc, mult, rep);

  __bf16* whi_i[5];
  int     fpad_i[5];
  {
    size_t woff = 0;
    for (int i = 0; i < 5; ++i) {
      int Fpad = (F[i] + 31) & ~31;
      fpad_i[i] = Fpad;
      whi_i[i]  = wsplit + woff;
      dim3 grid(Fpad / 32, H1 / 32), blk(32, 8);
      cvt_w1<<<grid, blk, 0, stream>>>(W1[i], whi_i[i],
                                       whi_i[i] + (size_t)H1 * Fpad, F[i], Fpad);
      woff += 2ull * H1 * Fpad;
    }
  }

  for (int i = 0; i < 5; ++i) {
    if (i > 0) {
      zero_pt<<<8, 256, 0, stream>>>(pt);
      int Kd = C[i - 1] * H2;                      // 4096..32768, /2048 exact
      prev_gemm<<<Kd / 2048, 256, 0, stream>>>(fp, W1[i] + (size_t)F[i] * H1,
                                               pt, Kd, 2048);
    }
    // strips = B*C/16 = C[i]; 4 waves per block
    gemm1_ss<<<C[i] / 4, 128, 0, stream>>>(P[i], whi_i[i], zraw,
                                           sfacc + i * 16, C[i], F[i], fpad_i[i]);
    finalize_sf<<<1, 32, 0, stream>>>(sfacc + i * 16, sfinv + i * 16, mult);
    stage2<<<C[i] / 4, 128, 0, stream>>>(zraw, B1[i], W2[i], B2[i], pt,
                                         sfinv + i * 16, rep + i * NB * H2,
                                         fp, C[i], i > 0 ? 1 : 0);
  }
  final_out<<<1, 32, 0, stream>>>(rep, mult, Wr, br, out);
}
